// MultiHeadAttention_16106127360563
// MI455X (gfx1250) — matrix operand order repaired and checked by reading, not compiled, not run on hardware
//
#include <hip/hip_runtime.h>
#include <stdint.h>

typedef __attribute__((ext_vector_type(16))) _Float16 v16h;
typedef __attribute__((ext_vector_type(8)))  float    v8f;

#define D_MODEL  1024
#define N_HEADS  16
#define HEAD_DIM 64
#define BATCH    2
#define SEQ      2048

// ---------------------------------------------------------------------------
// CDNA5 async copy: global -> LDS, 16 bytes per lane, tracked by ASYNCcnt.
// ---------------------------------------------------------------------------
static __device__ __forceinline__ uint32_t lds_off(const void* p) {
  // low 32 bits of a flat shared-aperture address are the LDS byte offset
  return (uint32_t)(uintptr_t)p;
}
static __device__ __forceinline__ void async_copy16(uint32_t lds, const void* g) {
  asm volatile("global_load_async_to_lds_b128 %0, %1, off"
               :: "v"(lds), "v"((uint64_t)(uintptr_t)g)
               : "memory");
}
template <int N>
static __device__ __forceinline__ void wait_async() {
  asm volatile("s_wait_asynccnt %0" :: "i"(N) : "memory");
}

// ---------------------------------------------------------------------------
// WMMA fragment loaders, contiguous 16-bit k order (wave32), same for A and B.
// A (16x32): lane%16 -> M; lane/16 selects K half {0-15}/{16-31}.
// B (32x16): lane%16 -> N; lane/16 selects K half {0-15}/{16-31}.
// Sources are row-major [row][k] arrays (LDS or global), rowstride in elems.
// ---------------------------------------------------------------------------
static __device__ __forceinline__ v16h load_a_frag(const _Float16* base, int rowstride,
                                                   int m0, int lane, int k0) {
  const int M = lane & 15, half = (lane >> 4) & 1;
  const _Float16* row = base + (size_t)(m0 + M) * rowstride + k0;
  v16h f;
  uint32_t* u = reinterpret_cast<uint32_t*>(&f);
#pragma unroll
  for (int v = 0; v < 4; ++v)
    u[v] = *reinterpret_cast<const uint32_t*>(row + 16 * half + 2 * v);
#pragma unroll
  for (int v = 0; v < 4; ++v)
    u[4 + v] = *reinterpret_cast<const uint32_t*>(row + 16 * half + 8 + 2 * v);
  return f;
}

static __device__ __forceinline__ v16h load_b_frag(const _Float16* base, int rowstride,
                                                   int n0, int lane, int k0) {
  const int N = lane & 15, half = (lane >> 4) & 1;
  const _Float16* row = base + (size_t)(n0 + N) * rowstride + k0;
  v16h f;
  uint32_t* u = reinterpret_cast<uint32_t*>(&f);
#pragma unroll
  for (int v = 0; v < 8; ++v)
    u[v] = *reinterpret_cast<const uint32_t*>(row + 16 * half + 2 * v);
  return f;
}

static __device__ __forceinline__ v8f wmma16(v16h a, v16h b, v8f c) {
  return __builtin_amdgcn_wmma_f32_16x16x32_f16(false, a, false, b, (short)0, c,
                                                false, false);
}

static __device__ __forceinline__ float rowmax16(float x) {
#pragma unroll
  for (int m = 8; m >= 1; m >>= 1) x = fmaxf(x, __shfl_xor(x, m, 16));
  return x;
}
static __device__ __forceinline__ float rowsum16(float x) {
#pragma unroll
  for (int m = 8; m >= 1; m >>= 1) x += __shfl_xor(x, m, 16);
  return x;
}

// ---------------------------------------------------------------------------
// One-time fp32 -> f16 conversion (x), and convert+transpose for weights:
// Wt[n][k] = W[k][n], exactly the layout the B-fragment loader consumes, so
// GEMM tile staging becomes a raw async row-slab copy.
// ---------------------------------------------------------------------------
__global__ __launch_bounds__(256) void cvt_f16_kernel(const float* __restrict__ src,
                                                      _Float16* __restrict__ dst,
                                                      int n4) {
  const int i = blockIdx.x * 256 + threadIdx.x;
  if (i >= n4) return;
  const float4 f = reinterpret_cast<const float4*>(src)[i];
  _Float16 h4[4] = {(_Float16)f.x, (_Float16)f.y, (_Float16)f.z, (_Float16)f.w};
  *reinterpret_cast<uint2*>(dst + 4 * (size_t)i) = *reinterpret_cast<uint2*>(h4);
}

__global__ __launch_bounds__(256) void cvt_transpose_kernel(const float* __restrict__ W,
                                                            _Float16* __restrict__ Wt) {
  const int i = blockIdx.x * 256 + threadIdx.x;  // 0 .. 1024*1024/4-1
  const int k = i >> 8;                          // 256 float4 per k-row
  const int n = (i & 255) * 4;
  const float4 f = *reinterpret_cast<const float4*>(W + (size_t)k * D_MODEL + n);
  Wt[(size_t)(n + 0) * D_MODEL + k] = (_Float16)f.x;
  Wt[(size_t)(n + 1) * D_MODEL + k] = (_Float16)f.y;
  Wt[(size_t)(n + 2) * D_MODEL + k] = (_Float16)f.z;
  Wt[(size_t)(n + 3) * D_MODEL + k] = (_Float16)f.w;
}

// ---------------------------------------------------------------------------
// GEMM + bias: out[M,N] = A[M,K] @ Wt[N,K]^T + bias, all-f16 inputs, fp32
// accumulate.  Block 256 (8 waves), tile 128x128x64, double-buffered LDS
// filled with async-to-LDS copies (8 x b128 per thread per tile).
// Wave w: rows [(w&3)*32,+32), cols [(w>>2)*64,+64) -> 2x4 wmma accums.
// VT: store f16 output per-head transposed [B,H,Hd,S] (for the V projection).
// ---------------------------------------------------------------------------
template <bool OUT_F32, bool VT>
__global__ __launch_bounds__(256) void gemm_bias_kernel(
    const _Float16* __restrict__ A, const _Float16* __restrict__ Wt,
    const float* __restrict__ bias, void* __restrict__ outp,
    int M, int N, int K) {
  __shared__ __attribute__((aligned(16))) _Float16 As[2][128][72];
  __shared__ __attribute__((aligned(16))) _Float16 Bs[2][128][72];

  const int t = threadIdx.x;
  const int lane = t & 31;
  const int wave = t >> 5;
  const int wm = wave & 3;
  const int wn = wave >> 2;
  const int m0blk = blockIdx.y * 128;
  const int n0blk = blockIdx.x * 128;

  const int srow = t >> 3;            // staging: 8 x b128 chunks per row
  const int scol = (t & 7) * 8;       // f16 element offset of 16B chunk

  v8f acc[2][4];
#pragma unroll
  for (int i = 0; i < 2; ++i)
#pragma unroll
    for (int j = 0; j < 4; ++j) acc[i][j] = (v8f){};

  const int NT = K >> 6;

  // stage(j): copy A[128x64] and Wt[128x64] k-slab j into buffer j&1
#define GEMM_STAGE(j)                                                          \
  do {                                                                         \
    const int kk0 = (j) << 6;                                                  \
    const int bf = (j) & 1;                                                    \
    _Pragma("unroll") for (int i = 0; i < 4; ++i) {                            \
      const int r = srow + 32 * i;                                             \
      async_copy16(lds_off(&As[bf][r][scol]),                                  \
                   A + (size_t)(m0blk + r) * K + kk0 + scol);                  \
      async_copy16(lds_off(&Bs[bf][r][scol]),                                  \
                   Wt + (size_t)(n0blk + r) * K + kk0 + scol);                 \
    }                                                                          \
  } while (0)

  GEMM_STAGE(0);
  for (int kt = 0; kt < NT; ++kt) {
    if (kt + 1 < NT) {
      GEMM_STAGE(kt + 1);
      wait_async<8>();   // tile kt's 8 copies done (async loads retire in order)
    } else {
      wait_async<0>();
    }
    __syncthreads();

    const int bf = kt & 1;
#pragma unroll
    for (int ks = 0; ks < 2; ++ks) {
      const v16h a0 = load_a_frag(&As[bf][0][0], 72, wm * 32, lane, ks * 32);
      const v16h a1 = load_a_frag(&As[bf][0][0], 72, wm * 32 + 16, lane, ks * 32);
#pragma unroll
      for (int nf = 0; nf < 4; ++nf) {
        const v16h b = load_b_frag(&Bs[bf][0][0], 72, wn * 64 + nf * 16, lane, ks * 32);
        acc[0][nf] = wmma16(a0, b, acc[0][nf]);
        acc[1][nf] = wmma16(a1, b, acc[1][nf]);
      }
    }
    __syncthreads();
  }
#undef GEMM_STAGE

  // epilogue (C/D layout: col = lane%16, row = v + 8*(lane/16))
  const int half = (lane >> 4) & 1;
  const int ln = lane & 15;
#pragma unroll
  for (int mf = 0; mf < 2; ++mf)
#pragma unroll
    for (int nf = 0; nf < 4; ++nf) {
      const int col = n0blk + wn * 64 + nf * 16 + ln;
      const float bv = bias[col];
#pragma unroll
      for (int v = 0; v < 8; ++v) {
        const int row = m0blk + wm * 32 + mf * 16 + v + 8 * half;
        const float val = acc[mf][nf][v] + bv;
        if (OUT_F32) {
          reinterpret_cast<float*>(outp)[(size_t)row * N + col] = val;
        } else if (VT) {
          const int bb = row >> 11;        // / SEQ
          const int s = row & (SEQ - 1);
          const int hh = col >> 6;
          const int hd = col & (HEAD_DIM - 1);
          reinterpret_cast<_Float16*>(outp)
              [(((size_t)bb * N_HEADS + hh) * HEAD_DIM + hd) * SEQ + s] = (_Float16)val;
        } else {
          reinterpret_cast<_Float16*>(outp)[(size_t)row * N + col] = (_Float16)val;
        }
      }
    }
}

// ---------------------------------------------------------------------------
// Flash attention, causal.  Grid (S/64, H, B), block 128 (4 waves); each wave
// owns 16 query rows.  K tiles ([key][hd] slab of Kh) and V tiles ([hd][key]
// slab of the pre-transposed Vt) staged by double-buffered async copies.
// Online softmax with lane-16 shuffles; P round-trips through per-wave LDS
// to reach the A-matrix layout for the P@V WMMA.
// ---------------------------------------------------------------------------
__global__ __launch_bounds__(128) void attention_kernel(
    const _Float16* __restrict__ Q, const _Float16* __restrict__ Km,
    const _Float16* __restrict__ Vt, _Float16* __restrict__ O) {
  __shared__ __attribute__((aligned(16))) _Float16 Ks[2][64][72];  // [key][hd]
  __shared__ __attribute__((aligned(16))) _Float16 Vs[2][64][72];  // [hd][key]
  __shared__ __attribute__((aligned(16))) _Float16 Ps[4][16][72];  // per-wave P

  const int t = threadIdx.x;
  const int lane = t & 31;
  const int wave = t >> 5;
  const int qt = blockIdx.x;
  const int h = blockIdx.y;
  const int b = blockIdx.z;
  const int half = (lane >> 4) & 1;
  const int ln = lane & 15;

  const size_t baseBH = (size_t)b * SEQ * D_MODEL + (size_t)h * HEAD_DIM;
  const _Float16* vbase = Vt + ((size_t)(b * N_HEADS + h) * HEAD_DIM) * SEQ;
  const int qrow0 = qt * 64 + wave * 16;

  // Q fragments live in registers for the whole key loop
  const _Float16* qptr = Q + baseBH + (size_t)qrow0 * D_MODEL;
  v16h qf[2];
  qf[0] = load_a_frag(qptr, D_MODEL, 0, lane, 0);
  qf[1] = load_a_frag(qptr, D_MODEL, 0, lane, 32);

  v8f o[4];
#pragma unroll
  for (int nf = 0; nf < 4; ++nf) o[nf] = (v8f){};
  float m8[8], l8[8];
#pragma unroll
  for (int v = 0; v < 8; ++v) { m8[v] = -__builtin_inff(); l8[v] = 0.0f; }

  const int srow = t >> 3;          // 0..15 base row, 8 chunks per row
  const int scol = (t & 7) * 8;     // 16B chunk offset in f16 elems
  const int NT = qt + 1;

  // stage(j): K tile rows are keys (stride D_MODEL); V tile rows are hd
  // (stride SEQ in the per-head transposed layout) -> both raw slab copies.
#define ATTN_STAGE(j)                                                          \
  do {                                                                         \
    const int kb = (j) << 6;                                                   \
    const int bf = (j) & 1;                                                    \
    _Pragma("unroll") for (int i = 0; i < 4; ++i) {                            \
      const int r = srow + 16 * i;                                             \
      async_copy16(lds_off(&Ks[bf][r][scol]),                                  \
                   Km + baseBH + (size_t)(kb + r) * D_MODEL + scol);           \
      async_copy16(lds_off(&Vs[bf][r][scol]),                                  \
                   vbase + (size_t)r * SEQ + kb + scol);                       \
    }                                                                          \
  } while (0)

  ATTN_STAGE(0);
  for (int kt = 0; kt < NT; ++kt) {
    if (kt + 1 < NT) {
      ATTN_STAGE(kt + 1);
      wait_async<8>();
    } else {
      wait_async<0>();
    }
    __syncthreads();
    const int bf = kt & 1;
    const int kbase = kt * 64;

    // scores S = (Q K^T) / sqrt(Hd)
    v8f s[4];
#pragma unroll
    for (int nf = 0; nf < 4; ++nf) {
      v8f a = (v8f){};
      a = wmma16(qf[0], load_b_frag(&Ks[bf][0][0], 72, nf * 16, lane, 0), a);
      a = wmma16(qf[1], load_b_frag(&Ks[bf][0][0], 72, nf * 16, lane, 32), a);
      s[nf] = a;
    }
    const bool diag = (kt == qt);
#pragma unroll
    for (int nf = 0; nf < 4; ++nf)
#pragma unroll
      for (int v = 0; v < 8; ++v) {
        float val = s[nf][v] * 0.125f;
        if (diag) {
          const int colg = kbase + nf * 16 + ln;
          const int rowg = qrow0 + v + 8 * half;
          if (colg > rowg) val = -1.0e30f;
        }
        s[nf][v] = val;
      }

    // online softmax (rows live across 16-lane halves)
    float mnew[8], resc[8];
#pragma unroll
    for (int v = 0; v < 8; ++v) {
      float cm = fmaxf(fmaxf(s[0][v], s[1][v]), fmaxf(s[2][v], s[3][v]));
      cm = rowmax16(cm);
      const float mn = fmaxf(m8[v], cm);
      mnew[v] = mn;
      resc[v] = __expf(m8[v] - mn);
    }
#pragma unroll
    for (int v = 0; v < 8; ++v) {
      float rs = 0.0f;
#pragma unroll
      for (int nf = 0; nf < 4; ++nf) {
        const float p = __expf(s[nf][v] - mnew[v]);
        s[nf][v] = p;
        rs += p;
      }
      rs = rowsum16(rs);
      l8[v] = l8[v] * resc[v] + rs;
      m8[v] = mnew[v];
    }
    // spill P (C/D layout) to LDS, reload in A layout (same-wave DS in order)
#pragma unroll
    for (int nf = 0; nf < 4; ++nf)
#pragma unroll
      for (int v = 0; v < 8; ++v)
        Ps[wave][v + 8 * half][nf * 16 + ln] = (_Float16)s[nf][v];
#pragma unroll
    for (int nf = 0; nf < 4; ++nf)
#pragma unroll
      for (int v = 0; v < 8; ++v) o[nf][v] *= resc[v];
#pragma unroll
    for (int ks = 0; ks < 2; ++ks) {
      const v16h pa = load_a_frag(&Ps[wave][0][0], 72, 0, lane, ks * 32);
#pragma unroll
      for (int nf = 0; nf < 4; ++nf) {
        const v16h vb = load_b_frag(&Vs[bf][0][0], 72, nf * 16, lane, ks * 32);
        o[nf] = wmma16(pa, vb, o[nf]);
      }
    }
    __syncthreads();
  }
#undef ATTN_STAGE

  // normalize and store context (f16, [B,S,D] with col = h*64+hd)
  float inv[8];
#pragma unroll
  for (int v = 0; v < 8; ++v) inv[v] = 1.0f / l8[v];
#pragma unroll
  for (int nf = 0; nf < 4; ++nf)
#pragma unroll
    for (int v = 0; v < 8; ++v) {
      const int rowg = qrow0 + v + 8 * half;
      const int colg = nf * 16 + ln;
      O[baseBH + (size_t)rowg * D_MODEL + colg] = (_Float16)(o[nf][v] * inv[v]);
    }
}

// ---------------------------------------------------------------------------
extern "C" void kernel_launch(void* const* d_in, const int* in_sizes, int n_in,
                              void* d_out, int out_size, void* d_ws, size_t ws_size,
                              hipStream_t stream) {
  (void)in_sizes; (void)n_in; (void)out_size; (void)ws_size;
  const float* x  = (const float*)d_in[0];
  const float* Wq = (const float*)d_in[1];
  const float* bq = (const float*)d_in[2];
  const float* Wk = (const float*)d_in[3];
  const float* bk = (const float*)d_in[4];
  const float* Wv = (const float*)d_in[5];
  const float* bv = (const float*)d_in[6];
  const float* Wo = (const float*)d_in[7];
  const float* bo = (const float*)d_in[8];

  const int M = BATCH * SEQ;   // 4096
  const int N = D_MODEL;       // 1024
  const int K = D_MODEL;       // 1024

  const size_t P = (size_t)M * D_MODEL;        // 4M elems per [B,S,D] plane
  const size_t WSZ = (size_t)D_MODEL * D_MODEL;
  _Float16* xh  = (_Float16*)d_ws;             // f16 workspace, ~48 MB total
  _Float16* Wqt = xh + P;
  _Float16* Wkt = Wqt + WSZ;
  _Float16* Wvt = Wkt + WSZ;
  _Float16* Wot = Wvt + WSZ;
  _Float16* Qh  = Wot + WSZ;
  _Float16* Kh  = Qh + P;
  _Float16* Vh  = Kh + P;    // per-head transposed [B,H,Hd,S]
  _Float16* Ch  = Vh + P;    // attention context

  cvt_f16_kernel<<<(int)(P / 4 / 256), 256, 0, stream>>>(x, xh, (int)(P / 4));
  const int wblk = (int)(WSZ / 4 / 256);
  cvt_transpose_kernel<<<wblk, 256, 0, stream>>>(Wq, Wqt);
  cvt_transpose_kernel<<<wblk, 256, 0, stream>>>(Wk, Wkt);
  cvt_transpose_kernel<<<wblk, 256, 0, stream>>>(Wv, Wvt);
  cvt_transpose_kernel<<<wblk, 256, 0, stream>>>(Wo, Wot);

  dim3 gg(N / 128, M / 128);                   // (8, 32)
  gemm_bias_kernel<false, false><<<gg, 256, 0, stream>>>(xh, Wqt, bq, (void*)Qh, M, N, K);
  gemm_bias_kernel<false, false><<<gg, 256, 0, stream>>>(xh, Wkt, bk, (void*)Kh, M, N, K);
  gemm_bias_kernel<false, true ><<<gg, 256, 0, stream>>>(xh, Wvt, bv, (void*)Vh, M, N, K);

  dim3 ga(SEQ / 64, N_HEADS, BATCH);           // (32, 16, 2)
  attention_kernel<<<ga, 128, 0, stream>>>(Qh, Kh, Vh, Ch);

  gemm_bias_kernel<true, false><<<gg, 256, 0, stream>>>(Ch, Wot, bo, d_out, M, N, K);
}
